// ModelSeR_6554120094233
// MI455X (gfx1250) — compile-verified, matrix operand order checked by reading, not run on hardware
//
#include <hip/hip_runtime.h>
#include <hip/hip_bf16.h>
#include <math.h>

// ---------------------------------------------------------------------------
// DeePMD se_r energy+force on gfx1250 (MI455X). wave32, WMMA f32_16x16x32_f16.
//
// WMMA register layouts (CDNA5 ISA 7.12.2, wave32):
//   A (16x32 f16): lane l -> row m=l&15, half h=l>>4; element e of v16h holds
//                  K = ((e>>3)<<4) + h*8 + (e&7).
//   B (32x16 f16): mirror of A with lane = N column;
//                  same K(e) formula, n=l&15.
//   C/D (16x16 f32): lane l -> n=l&15; VGPR v -> m = v + 8*(l>>4).
// ---------------------------------------------------------------------------

typedef __attribute__((ext_vector_type(16))) _Float16 v16h;
typedef __attribute__((ext_vector_type(8)))  float    v8f;

#define RCUT      6.0f
#define RCUT_SMTH 5.8f
#define NNEI      96
#define NBATCH    4
#define NATOM     1024
#define NA        (NBATCH * NATOM)      // 4096 atoms total
#define MROWS     ((size_t)NA * NNEI)   // 393216 embedding rows
#define G_SCALE   (0.2f / 96.0f)

// ---- workspace layout (byte offsets) --------------------------------------
#define OFF_S     ((size_t)0)
#define OFF_DSDR  ((size_t)1572864)
#define OFF_NBR   ((size_t)3145728)
#define OFF_G16   ((size_t)4718592)     // [NA][128] f16 (K-pad for fit1)
#define OFF_T1    ((size_t)5767168)     // [NA][256] f16 tanh1 (later u1)
#define OFF_T2    ((size_t)7864320)     // [NA][256] f16 tanh2 (later u2)
#define OFF_Y2    ((size_t)9961472)     // [NA][256] f16 y2
#define OFF_T3    ((size_t)12058624)    // [NA][256] f16 tanh3 (later u3)
#define OFF_Y3    ((size_t)14155776)    // [NA][256] f16 y3
#define OFF_DY2   ((size_t)16252928)    // [NA][256] f16
#define OFF_DY1   ((size_t)18350080)    // [NA][256] f16
#define OFF_DG    ((size_t)20447232)    // [NA][112] f32 (scaled dE/dh2)
#define OFF_END   ((size_t)22282240)

// Hardware tanh (CDNA5 V_TANH_F32 is a TRANS32 op); branchless fallback via
// exp2/rcp: tanh(x) = 1 - 2/(exp2(2x*log2e)+1)  (exact limits at +-inf).
__device__ __forceinline__ float fast_tanh(float x) {
#if __has_builtin(__builtin_amdgcn_tanhf)
  return __builtin_amdgcn_tanhf(x);
#elif __has_builtin(__builtin_amdgcn_tanh_f32)
  return __builtin_amdgcn_tanh_f32(x);
#else
  float e = __builtin_amdgcn_exp2f(2.8853900817779268f * x);
  return 1.f - 2.f * __builtin_amdgcn_rcpf(e + 1.f);
#endif
}

__device__ __forceinline__ v8f wmma16(v16h a, v16h b, v8f c) {
  return __builtin_amdgcn_wmma_f32_16x16x32_f16(
      /*neg_a=*/false, a, /*neg_b=*/false, b,
      /*c_mod=*/(short)0, c, /*reuse_a=*/false, /*reuse_b=*/false);
}

// A fragment from a row-major 16 x ld tile (LDS or global); K pads must be 0.
__device__ __forceinline__ v16h fragA(const _Float16* base, int ld, int kbase, int lane) {
  int m  = lane & 15;
  int h8 = (lane >> 4) * 8;
  v16h a;
#pragma unroll
  for (int e = 0; e < 16; ++e) {
    int k = kbase + ((e >> 3) << 4) + h8 + (e & 7);
    a[e] = base[m * ld + k];
  }
  return a;
}

// B fragment from row-major f16 [K x N] (leading dim ld), zero beyond kmax.
__device__ __forceinline__ v16h fragB16(const _Float16* base, int ld, int kbase,
                                        int nbase, int kmax, int lane) {
  int n  = nbase + (lane & 15);
  int h8 = (lane >> 4) * 8;
  v16h b;
#pragma unroll
  for (int e = 0; e < 16; ++e) {
    int k = kbase + ((e >> 3) << 4) + h8 + (e & 7);
    b[e] = (k < kmax) ? base[k * ld + n] : (_Float16)0.f;
  }
  return b;
}

// B fragment of W^T: B[k][n] = base[n*ld + k].
__device__ __forceinline__ v16h fragB16T(const _Float16* base, int ld, int kbase,
                                         int nbase, int kmax, int lane) {
  int n  = nbase + (lane & 15);
  int h8 = (lane >> 4) * 8;
  v16h b;
#pragma unroll
  for (int e = 0; e < 16; ++e) {
    int k = kbase + ((e >> 3) << 4) + h8 + (e & 7);
    b[e] = (k < kmax) ? base[n * ld + k] : (_Float16)0.f;
  }
  return b;
}

// B fragment from global f32 weights, optional transpose, full bound guards.
__device__ __forceinline__ v16h fragBg(const float* W, int ldw, int transW,
                                       int kbase, int nbase, int K, int N, int lane) {
  int n  = nbase + (lane & 15);
  int h8 = (lane >> 4) * 8;
  v16h b;
#pragma unroll
  for (int e = 0; e < 16; ++e) {
    int k = kbase + ((e >> 3) << 4) + h8 + (e & 7);
    float v = 0.f;
    if (k < K && n < N) v = transW ? W[(size_t)n * ldw + k] : W[(size_t)k * ldw + n];
    b[e] = (_Float16)v;
  }
  return b;
}

// ---------------------------------------------------------------------------
// 0) zero the output + f16/f32 workspace region (pads must be zero each call)
// ---------------------------------------------------------------------------
__global__ void k_zero(float* out, unsigned nout, unsigned* ws, size_t nws) {
  size_t i = (size_t)blockIdx.x * blockDim.x + threadIdx.x;
  size_t stride = (size_t)gridDim.x * blockDim.x;
  for (size_t t = i; t < (size_t)nout; t += stride) out[t] = 0.f;
  for (size_t t = i; t < nws; t += stride) ws[t] = 0u;
}

// ---------------------------------------------------------------------------
// 1) neighbor search: per-atom block, full bitonic sort of 1024 squared dists
// ---------------------------------------------------------------------------
__global__ __launch_bounds__(256) void k_neighbors(const float* __restrict__ coord,
                                                   float* __restrict__ s,
                                                   float* __restrict__ dsdr,
                                                   int* __restrict__ nbr) {
  __shared__ float kd[NATOM];
  __shared__ int   ki[NATOM];
  int a = blockIdx.x;                 // global atom id
  int b = a >> 10, i = a & 1023;
  const float* cb = coord + (size_t)b * NATOM * 3;
  float xi = cb[i * 3 + 0], yi = cb[i * 3 + 1], zi = cb[i * 3 + 2];
  for (int j = threadIdx.x; j < NATOM; j += blockDim.x) {
    float dx = xi - cb[j * 3 + 0];
    float dy = yi - cb[j * 3 + 1];
    float dz = zi - cb[j * 3 + 2];
    float d2 = dx * dx + dy * dy + dz * dz;
    kd[j] = (j == i) ? 1e30f : d2;
    ki[j] = j;
  }
  __syncthreads();
  for (unsigned k = 2; k <= NATOM; k <<= 1) {
    for (unsigned j = k >> 1; j > 0; j >>= 1) {
      for (unsigned t = threadIdx.x; t < NATOM; t += blockDim.x) {
        unsigned ixj = t ^ j;
        if (ixj > t) {
          bool up = ((t & k) == 0);
          float a0 = kd[t], a1 = kd[ixj];
          if ((a0 > a1) == up) {
            kd[t] = a1; kd[ixj] = a0;
            int tmp = ki[t]; ki[t] = ki[ixj]; ki[ixj] = tmp;
          }
        }
      }
      __syncthreads();
    }
  }
  if (threadIdx.x < NNEI) {
    int t = threadIdx.x;
    float r = sqrtf(kd[t]);
    float sv = 0.f, dv = 0.f;
    if (r < RCUT) {
      float rr = fmaxf(r, 1e-6f);
      if (r < RCUT_SMTH) {
        sv = 1.f / rr;
        dv = -1.f / (rr * rr);
      } else {
        float uu = (r - RCUT_SMTH) / (RCUT - RCUT_SMTH);
        float vv = uu * uu * uu * (-6.f * uu * uu + 15.f * uu - 10.f) + 1.f;
        float dvv = (-30.f * uu * uu * (uu - 1.f) * (uu - 1.f)) / (RCUT - RCUT_SMTH);
        sv = vv / rr;
        dv = dvv / rr - vv / (rr * rr);
      }
    }
    size_t o = (size_t)a * NNEI + t;
    s[o] = sv; dsdr[o] = dv; nbr[o] = ki[t];
  }
}

// ---------------------------------------------------------------------------
// 2) embedding forward: wave-per-atom, 6 WMMA row-tiles, neighbor-mean -> g
// ---------------------------------------------------------------------------
#define EF_WAVES 8
__global__ __launch_bounds__(256) void k_embed_fwd(
    const float* __restrict__ sbuf,
    const float* __restrict__ fw0, const float* __restrict__ fb0,
    const float* __restrict__ fw1, const float* __restrict__ fb1,
    const float* __restrict__ fw2, const float* __restrict__ fb2,
    _Float16* __restrict__ g16) {
  __shared__ _Float16 wf1s[32 * 64];       // fw1 padded [32][64]
  __shared__ _Float16 wf2s[64 * 112];      // fw2 padded [64][112]
  __shared__ _Float16 H0s[EF_WAVES * 16 * 32];
  __shared__ _Float16 H1s[EF_WAVES * 16 * 64];
  for (int idx = threadIdx.x; idx < 32 * 64; idx += blockDim.x) {
    int k = idx >> 6, n = idx & 63;
    wf1s[idx] = (k < 25 && n < 50) ? (_Float16)fw1[k * 50 + n] : (_Float16)0.f;
  }
  for (int idx = threadIdx.x; idx < 64 * 112; idx += blockDim.x) {
    int k = idx / 112, n = idx % 112;
    wf2s[idx] = (k < 50 && n < 100) ? (_Float16)fw2[k * 100 + n] : (_Float16)0.f;
  }
  __syncthreads();

  int lane = threadIdx.x & 31, wv = threadIdx.x >> 5;
  int atom = blockIdx.x * EF_WAVES + wv;
  _Float16* H0 = H0s + wv * 16 * 32;
  _Float16* H1 = H1s + wv * 16 * 64;
  int m = lane & 15, hb = lane >> 4, nl = lane & 15;
  float colsum[7];
#pragma unroll
  for (int t = 0; t < 7; ++t) colsum[t] = 0.f;

  for (int rt = 0; rt < 6; ++rt) {
    int rowbase = atom * NNEI + rt * 16;
    float x = sbuf[rowbase + m];
#pragma unroll
    for (int cc = 0; cc < 16; ++cc) {      // h0 = tanh(x*fw0 + fb0), pad->0
      int c = hb * 16 + cc;
      int cs = c < 25 ? c : 0;
      float hv = fast_tanh(x * fw0[cs] + fb0[cs]);
      H0[m * 32 + c] = (c < 25) ? (_Float16)hv : (_Float16)0.f;
    }
    v16h a0 = fragA(H0, 32, 0, lane);
    for (int nt = 0; nt < 4; ++nt) {       // z1 = h0 @ fw1
      v8f acc = {};
      acc = wmma16(a0, fragB16(wf1s, 64, 0, nt * 16, 32, lane), acc);
#pragma unroll
      for (int v = 0; v < 8; ++v) {        // h1 = dup(h0) + tanh(z1+fb1)
        int mm = v + hb * 8, c = nt * 16 + nl;
        int cs = c < 50 ? c : 0;
        float h0d = (float)H0[mm * 32 + (cs < 25 ? cs : cs - 25)];
        float hv = h0d + fast_tanh(acc[v] + fb1[cs]);
        H1[mm * 64 + c] = (c < 50) ? (_Float16)hv : (_Float16)0.f;
      }
    }
    for (int nt = 0; nt < 7; ++nt) {       // z2 = h1 @ fw2 ; accumulate mean
      v8f acc = {};
      for (int kt = 0; kt < 2; ++kt)
        acc = wmma16(fragA(H1, 64, kt * 32, lane),
                     fragB16(wf2s, 112, kt * 32, nt * 16, 64, lane), acc);
      float ps = 0.f;
#pragma unroll
      for (int v = 0; v < 8; ++v) {
        int mm = v + hb * 8, c = nt * 16 + nl;
        int cs = c < 100 ? c : 0;
        float h1d = (float)H1[mm * 64 + (cs < 50 ? cs : cs - 50)];
        float h2 = h1d + fast_tanh(acc[v] + fb2[cs]);
        ps += (c < 100) ? h2 : 0.f;
      }
      ps += __shfl_xor(ps, 16, 32);        // fold the two m-halves
      colsum[nt] += ps;
    }
  }
  if (lane < 16) {
#pragma unroll
    for (int nt = 0; nt < 7; ++nt) {
      int c = nt * 16 + lane;
      float val = (c < 100) ? colsum[nt] * G_SCALE : 0.f;
      g16[(size_t)atom * 128 + c] = (_Float16)val;
    }
  }
}

// ---------------------------------------------------------------------------
// 3) generic WMMA GEMM for the fitting net (fwd + bwd), wave-per-16-row tile
//    mode 0: out = tanh(z + bias)
//    mode 1: t = tanh(z+bias); out2 = t; out = skip + t*idt
//    mode 2: out = z [+ bias] [+ skip]
//    mode 3: out32 = z * scale
// ---------------------------------------------------------------------------
__global__ __launch_bounds__(256) void k_gemm(
    const _Float16* __restrict__ A, int lda, int K,
    const float* __restrict__ W, int ldw, int transW,
    int M, int N,
    const float* __restrict__ bias,
    const _Float16* __restrict__ skip, int ldskip,
    const float* __restrict__ idt,
    _Float16* __restrict__ out16, int ldo,
    _Float16* __restrict__ out2,
    float* __restrict__ out32, int ldo32,
    float scale, int mode) {
  int lane = threadIdx.x & 31, wv = threadIdx.x >> 5;
  int rowtile = blockIdx.x * (blockDim.x >> 5) + wv;
  int row0 = rowtile * 16;
  if (row0 >= M) return;
  __builtin_prefetch(W, 0, 1);             // global_prefetch_b8 on weights
  int KT = (K + 31) >> 5;
  int NT = (N + 15) >> 4;
  const _Float16* Abase = A + (size_t)row0 * lda;
  v16h af[8];
  for (int kt = 0; kt < KT; ++kt) af[kt] = fragA(Abase, lda, kt * 32, lane);
  int h8 = (lane >> 4) * 8, nl = lane & 15;
  for (int nt = 0; nt < NT; ++nt) {
    v8f acc = {};
    for (int kt = 0; kt < KT; ++kt)
      acc = wmma16(af[kt], fragBg(W, ldw, transW, kt * 32, nt * 16, K, N, lane), acc);
    int n = nt * 16 + nl;
    if (n < N) {                           // store guard (lane-divergent, reconverges)
#pragma unroll
      for (int v = 0; v < 8; ++v) {
        int row = row0 + v + h8;
        size_t oi = (size_t)row * ldo + n;
        float z = acc[v];
        if (mode == 0) {
          out16[oi] = (_Float16)fast_tanh(z + bias[n]);
        } else if (mode == 1) {
          float t = fast_tanh(z + bias[n]);
          out2[oi] = (_Float16)t;
          out16[oi] = (_Float16)((float)skip[(size_t)row * ldskip + n] + t * idt[n]);
        } else if (mode == 2) {
          float o = z;
          if (bias) o += bias[n];
          if (skip) o += (float)skip[(size_t)row * ldskip + n];
          out16[oi] = (_Float16)o;
        } else {
          out32[(size_t)row * ldo32 + n] = z * scale;
        }
      }
    }
  }
}

// ---------------------------------------------------------------------------
// 4) energies: e_raw = y3 . wfin + bfin ; energy[b] += e_raw (atomic)
// ---------------------------------------------------------------------------
__global__ void k_final(const _Float16* __restrict__ y3,
                        const float* __restrict__ wfin,
                        const float* __restrict__ bfin,
                        float* __restrict__ out) {
  int a = blockIdx.x * blockDim.x + threadIdx.x;
  if (a >= NA) return;
  const _Float16* yr = y3 + (size_t)a * 256;
  float acc = bfin[0];
  for (int c = 0; c < 240; ++c) acc += (float)yr[c] * wfin[c];
  out[NBATCH + NBATCH * 3 * NATOM + a] = acc;   // energy_raw
  atomicAdd(&out[a >> 10], acc);                // per-batch energy
}

// ---------------------------------------------------------------------------
// 5) elementwise backward helpers (in-place into the tanh buffers)
// ---------------------------------------------------------------------------
__global__ void k_u3(_Float16* t3, const float* wfin, const float* idt3) {
  int idx = blockIdx.x * blockDim.x + threadIdx.x;
  if (idx >= NA * 240) return;
  int r = idx / 240, n = idx % 240;
  size_t o = (size_t)r * 256 + n;
  float t = (float)t3[o];
  t3[o] = (_Float16)(wfin[n] * idt3[n] * (1.f - t * t));
}
__global__ void k_u2(_Float16* t2, const _Float16* dy2, const float* idt2) {
  int idx = blockIdx.x * blockDim.x + threadIdx.x;
  if (idx >= NA * 240) return;
  int r = idx / 240, n = idx % 240;
  size_t o = (size_t)r * 256 + n;
  float t = (float)t2[o];
  t2[o] = (_Float16)((float)dy2[o] * idt2[n] * (1.f - t * t));
}
__global__ void k_u1(_Float16* t1, const _Float16* dy1) {
  int idx = blockIdx.x * blockDim.x + threadIdx.x;
  if (idx >= NA * 240) return;
  int r = idx / 240, n = idx % 240;
  size_t o = (size_t)r * 256 + n;
  float t = (float)t1[o];
  t1[o] = (_Float16)((float)dy1[o] * (1.f - t * t));
}

// ---------------------------------------------------------------------------
// 6) embedding backward: recompute fwd tiles, WMMA backward GEMMs against the
//    SAME staged weights (index-transposed gather), force scatter via atomics
// ---------------------------------------------------------------------------
#define EB_WAVES 2
__global__ __launch_bounds__(64) void k_embed_bwd(
    const float* __restrict__ coord,
    const float* __restrict__ sbuf, const float* __restrict__ dsdr,
    const int* __restrict__ nbr, const float* __restrict__ dgbuf,
    const float* __restrict__ fw0, const float* __restrict__ fb0,
    const float* __restrict__ fw1, const float* __restrict__ fb1,
    const float* __restrict__ fw2, const float* __restrict__ fb2,
    float* __restrict__ force) {
  __shared__ _Float16 wf1s[32 * 64];
  __shared__ _Float16 wf2s[64 * 112];
  __shared__ _Float16 H0s[EB_WAVES * 16 * 32];
  __shared__ _Float16 H1s[EB_WAVES * 16 * 64];
  __shared__ _Float16 DZ2s[EB_WAVES * 16 * 128];
  __shared__ _Float16 DZ1s[EB_WAVES * 16 * 64];
  __shared__ _Float16 DH1s[EB_WAVES * 16 * 64];
  __shared__ float    DH0s[EB_WAVES * 16 * 32];
  __shared__ float    DH2s[EB_WAVES * 112];
  for (int idx = threadIdx.x; idx < 32 * 64; idx += blockDim.x) {
    int k = idx >> 6, n = idx & 63;
    wf1s[idx] = (k < 25 && n < 50) ? (_Float16)fw1[k * 50 + n] : (_Float16)0.f;
  }
  for (int idx = threadIdx.x; idx < 64 * 112; idx += blockDim.x) {
    int k = idx / 112, n = idx % 112;
    wf2s[idx] = (k < 50 && n < 100) ? (_Float16)fw2[k * 100 + n] : (_Float16)0.f;
  }
  __syncthreads();

  int lane = threadIdx.x & 31, wv = threadIdx.x >> 5;
  int atom = blockIdx.x * EB_WAVES + wv;
  _Float16* H0  = H0s  + wv * 16 * 32;
  _Float16* H1  = H1s  + wv * 16 * 64;
  _Float16* DZ2 = DZ2s + wv * 16 * 128;
  _Float16* DZ1 = DZ1s + wv * 16 * 64;
  _Float16* DH1 = DH1s + wv * 16 * 64;
  float*    DH0 = DH0s + wv * 16 * 32;
  float*    dh2 = DH2s + wv * 112;
  int m = lane & 15, hb = lane >> 4, nl = lane & 15;

  for (int c = lane; c < 112; c += 32) dh2[c] = dgbuf[(size_t)atom * 112 + c];
  for (int idx = lane; idx < 256; idx += 32)                 // zero DZ2 K-pad
    DZ2[(idx >> 4) * 128 + 112 + (idx & 15)] = (_Float16)0.f;

  int b = atom >> 10, i = atom & 1023;
  const float* cb = coord + (size_t)b * NATOM * 3;
  float* fob = force + (size_t)b * NATOM * 3;

  for (int rt = 0; rt < 6; ++rt) {
    int rowbase = atom * NNEI + rt * 16;
    // ---- recompute forward tile ----
    float x = sbuf[rowbase + m];
#pragma unroll
    for (int cc = 0; cc < 16; ++cc) {
      int c = hb * 16 + cc;
      int cs = c < 25 ? c : 0;
      float hv = fast_tanh(x * fw0[cs] + fb0[cs]);
      H0[m * 32 + c] = (c < 25) ? (_Float16)hv : (_Float16)0.f;
    }
    v16h a0 = fragA(H0, 32, 0, lane);
    for (int nt = 0; nt < 4; ++nt) {
      v8f acc = {};
      acc = wmma16(a0, fragB16(wf1s, 64, 0, nt * 16, 32, lane), acc);
#pragma unroll
      for (int v = 0; v < 8; ++v) {
        int mm = v + hb * 8, c = nt * 16 + nl;
        int cs = c < 50 ? c : 0;
        float h0d = (float)H0[mm * 32 + (cs < 25 ? cs : cs - 25)];
        float hv = h0d + fast_tanh(acc[v] + fb1[cs]);
        H1[mm * 64 + c] = (c < 50) ? (_Float16)hv : (_Float16)0.f;
      }
    }
    // ---- layer2 forward -> dz2 = dh2 * (1 - tanh^2) ----
    for (int nt = 0; nt < 7; ++nt) {
      v8f acc = {};
      for (int kt = 0; kt < 2; ++kt)
        acc = wmma16(fragA(H1, 64, kt * 32, lane),
                     fragB16(wf2s, 112, kt * 32, nt * 16, 64, lane), acc);
#pragma unroll
      for (int v = 0; v < 8; ++v) {
        int mm = v + hb * 8, c = nt * 16 + nl;
        int cs = c < 100 ? c : 0;
        float tc = fast_tanh(acc[v] + fb2[cs]);
        float dz = dh2[cs] * (1.f - tc * tc);
        DZ2[mm * 128 + c] = (c < 100) ? (_Float16)dz : (_Float16)0.f;
      }
    }
    // ---- dh1 = fold50(dh2) + dz2 @ fw2^T ; dz1 = dh1*(1-tb^2) ----
    for (int nt = 0; nt < 4; ++nt) {
      v8f acc = {};
      for (int kt = 0; kt < 4; ++kt)
        acc = wmma16(fragA(DZ2, 128, kt * 32, lane),
                     fragB16T(wf2s, 112, kt * 32, nt * 16, 112, lane), acc);
#pragma unroll
      for (int v = 0; v < 8; ++v) {
        int mm = v + hb * 8, c = nt * 16 + nl;
        int cs = c < 50 ? c : 0;
        float dh1v = dh2[cs] + dh2[cs + 50] + acc[v];
        float h0d = (float)H0[mm * 32 + (cs < 25 ? cs : cs - 25)];
        float tb = (float)H1[mm * 64 + cs] - h0d;
        float dz1 = dh1v * (1.f - tb * tb);
        DZ1[mm * 64 + c] = (c < 50) ? (_Float16)dz1 : (_Float16)0.f;
        DH1[mm * 64 + c] = (c < 50) ? (_Float16)dh1v : (_Float16)0.f;
      }
    }
    // ---- dh0 = fold25(dh1) + dz1 @ fw1^T ; weight by (1-h0^2)*fw0 ----
    for (int nt = 0; nt < 2; ++nt) {
      v8f acc = {};
      for (int kt = 0; kt < 2; ++kt)
        acc = wmma16(fragA(DZ1, 64, kt * 32, lane),
                     fragB16T(wf1s, 64, kt * 32, nt * 16, 64, lane), acc);
#pragma unroll
      for (int v = 0; v < 8; ++v) {
        int mm = v + hb * 8, c = nt * 16 + nl;
        int cs = c < 25 ? c : 0;
        float dh0v = (float)DH1[mm * 64 + cs] + (float)DH1[mm * 64 + cs + 25] + acc[v];
        float h0 = (float)H0[mm * 32 + cs];
        float dval = dh0v * (1.f - h0 * h0) * fw0[cs];
        DH0[mm * 32 + c] = (c < 25) ? dval : 0.f;
      }
    }
    // ---- dx = sum_c DH0 ; scatter forces ----
    if (lane < 16) {
      float dx = 0.f;
      for (int c = 0; c < 25; ++c) dx += DH0[lane * 32 + c];
      int ridx = rowbase + lane;
      float dd = dsdr[ridx];
      if (dd != 0.f) {
        int j = nbr[ridx];
        float ddx = cb[i * 3 + 0] - cb[j * 3 + 0];
        float ddy = cb[i * 3 + 1] - cb[j * 3 + 1];
        float ddz = cb[i * 3 + 2] - cb[j * 3 + 2];
        float r = sqrtf(ddx * ddx + ddy * ddy + ddz * ddz);
        float f = dx * dd / fmaxf(r, 1e-12f);   // dE/dc_i along unit vec
        atomicAdd(&fob[i * 3 + 0], -f * ddx);   // force = -grad
        atomicAdd(&fob[i * 3 + 1], -f * ddy);
        atomicAdd(&fob[i * 3 + 2], -f * ddz);
        atomicAdd(&fob[j * 3 + 0],  f * ddx);
        atomicAdd(&fob[j * 3 + 1],  f * ddy);
        atomicAdd(&fob[j * 3 + 2],  f * ddz);
      }
    }
  }
}

// ---------------------------------------------------------------------------
extern "C" void kernel_launch(void* const* d_in, const int* in_sizes, int n_in,
                              void* d_out, int out_size, void* d_ws, size_t ws_size,
                              hipStream_t stream) {
  (void)in_sizes; (void)n_in; (void)out_size; (void)ws_size;
  const float* coord = (const float*)d_in[0];
  const float* fw0  = (const float*)d_in[2];
  const float* fb0  = (const float*)d_in[3];
  const float* fw1  = (const float*)d_in[4];
  const float* fb1  = (const float*)d_in[5];
  const float* fw2  = (const float*)d_in[6];
  const float* fb2  = (const float*)d_in[7];
  const float* nw1  = (const float*)d_in[8];
  const float* nb1  = (const float*)d_in[9];
  const float* nw2  = (const float*)d_in[10];
  const float* nb2  = (const float*)d_in[11];
  const float* idt2 = (const float*)d_in[12];
  const float* nw3  = (const float*)d_in[13];
  const float* nb3  = (const float*)d_in[14];
  const float* idt3 = (const float*)d_in[15];
  const float* wfin = (const float*)d_in[16];
  const float* bfin = (const float*)d_in[17];

  char* ws = (char*)d_ws;
  float* out = (float*)d_out;
  float*     s_p    = (float*)(ws + OFF_S);
  float*     dsdr_p = (float*)(ws + OFF_DSDR);
  int*       nbr_p  = (int*)(ws + OFF_NBR);
  _Float16*  g16    = (_Float16*)(ws + OFF_G16);
  _Float16*  t1     = (_Float16*)(ws + OFF_T1);
  _Float16*  t2     = (_Float16*)(ws + OFF_T2);
  _Float16*  y2     = (_Float16*)(ws + OFF_Y2);
  _Float16*  t3     = (_Float16*)(ws + OFF_T3);
  _Float16*  y3     = (_Float16*)(ws + OFF_Y3);
  _Float16*  dy2    = (_Float16*)(ws + OFF_DY2);
  _Float16*  dy1    = (_Float16*)(ws + OFF_DY1);
  float*     dg     = (float*)(ws + OFF_DG);

  // 0) zero outputs + padded workspace (energy/force accumulators, K-pads)
  k_zero<<<2048, 256, 0, stream>>>(out, 16388u,
                                   (unsigned*)(ws + OFF_G16),
                                   (OFF_END - OFF_G16) / 4);
  // 1) descriptor
  k_neighbors<<<NA, 256, 0, stream>>>(coord, s_p, dsdr_p, nbr_p);
  // 2) embedding forward -> g
  k_embed_fwd<<<NA / EF_WAVES, 256, 0, stream>>>(s_p, fw0, fb0, fw1, fb1, fw2, fb2, g16);
  // 3) fitting forward
  k_gemm<<<32, 256, 0, stream>>>(g16, 128, 100, nw1, 240, 0, NA, 240, nb1,
                                 nullptr, 0, nullptr, t1, 256, nullptr,
                                 nullptr, 0, 1.f, 0);
  k_gemm<<<32, 256, 0, stream>>>(t1, 256, 240, nw2, 240, 0, NA, 240, nb2,
                                 t1, 256, idt2, y2, 256, t2,
                                 nullptr, 0, 1.f, 1);
  k_gemm<<<32, 256, 0, stream>>>(y2, 256, 240, nw3, 240, 0, NA, 240, nb3,
                                 y2, 256, idt3, y3, 256, t3,
                                 nullptr, 0, 1.f, 1);
  k_final<<<16, 256, 0, stream>>>(y3, wfin, bfin, out);
  // 4) fitting backward: dy2 = wfin + u3@nw3^T ; dy1 = dy2 + u2@nw2^T ; dg
  k_u3<<<3840, 256, 0, stream>>>(t3, wfin, idt3);
  k_gemm<<<32, 256, 0, stream>>>(t3, 256, 240, nw3, 240, 1, NA, 240, wfin,
                                 nullptr, 0, nullptr, dy2, 256, nullptr,
                                 nullptr, 0, 1.f, 2);
  k_u2<<<3840, 256, 0, stream>>>(t2, dy2, idt2);
  k_gemm<<<32, 256, 0, stream>>>(t2, 256, 240, nw2, 240, 1, NA, 240, nullptr,
                                 dy2, 256, nullptr, dy1, 256, nullptr,
                                 nullptr, 0, 1.f, 2);
  k_u1<<<3840, 256, 0, stream>>>(t1, dy1);
  k_gemm<<<32, 256, 0, stream>>>(t1, 256, 240, nw1, 240, 1, NA, 100, nullptr,
                                 nullptr, 0, nullptr, nullptr, 0, nullptr,
                                 dg, 112, G_SCALE, 3);
  // 5) embedding backward + force scatter
  k_embed_bwd<<<NA / EB_WAVES, 64, 0, stream>>>(coord, s_p, dsdr_p, nbr_p, dg,
                                                fw0, fb0, fw1, fb1, fw2, fb2,
                                                out + NBATCH);
}
